// MHSelfAttention_5093831213592
// MI455X (gfx1250) — matrix operand order repaired and checked by reading, not compiled, not run on hardware
//
#include <hip/hip_runtime.h>
#include <hip/hip_bf16.h>

// MH self-attention forward for MI455X (gfx1250), wave32 + WMMA f16.
// Stages: f32->f16 convert | QKV GEMM (wmma) | flash attention (wmma) | out GEMM (wmma).
// This revision: __launch_bounds__(128,1) to kill scratch spills in the wmma
// loops, and ping-pong (X/Y) K-chunk buffers to remove register-rotation movs.

typedef __attribute__((ext_vector_type(16))) _Float16 v16h;
typedef __attribute__((ext_vector_type(8)))  float    v8f;
typedef __attribute__((ext_vector_type(4)))  float    f32x4;
typedef __attribute__((ext_vector_type(4)))  _Float16 h16x4;

#define B_   4
#define T_   2048
#define KD_  1024
#define H_   16
#define S_   64
#define QKV_ 3072

__device__ __forceinline__ v8f wmma_f16(v16h a, v16h b, v8f c) {
  // (neg_a, A, neg_b, B, c_mod, C, reuse_a, reuse_b)
  return __builtin_amdgcn_wmma_f32_16x16x32_f16(false, a, false, b, (short)0, c,
                                                false, false);
}

// A-matrix (16x32 f16): lane L holds row L%16; VGPR0-3 = K klo..klo+7,
// VGPR4-7 = K klo+16..klo+23, klo = (L<16 ? 0 : 8). Source is row-major.
__device__ __forceinline__ v16h load_A16(const _Float16* base, int stride, int lane) {
  const int row = lane & 15;
  const int klo = (lane & 16) ? 8 : 0;
  const _Float16* p = base + (size_t)row * stride + klo;
  v16h r;
  ((f32x4*)&r)[0] = *(const f32x4*)(p);
  ((f32x4*)&r)[1] = *(const f32x4*)(p + 16);
  return r;
}

// B-matrix (32x16 f16): lane L holds column N = L%16; VGPR0-3 = K klo..klo+7,
// VGPR4-7 = K klo+16..klo+23, klo = (L<16 ? 0 : 8). Source supplies column N as a
// contiguous row of B^T (row-major), i.e. base + N*stride + koff.
__device__ __forceinline__ v16h load_B16(const _Float16* base, int stride, int lane) {
  const _Float16* p = base + (size_t)(lane & 15) * stride + ((lane & 16) ? 8 : 0);
  v16h r;
  ((f32x4*)&r)[0] = *(const f32x4*)(p);
  ((f32x4*)&r)[1] = *(const f32x4*)(p + 16);
  return r;
}

__global__ void f32_to_f16_kernel(const float* __restrict__ in,
                                  _Float16* __restrict__ out, int n4) {
  int i = blockIdx.x * blockDim.x + threadIdx.x;
  if (i < n4) {
    f32x4 v = ((const f32x4*)in)[i];
    h16x4 h;
    h.x = (_Float16)v.x; h.y = (_Float16)v.y;
    h.z = (_Float16)v.z; h.w = (_Float16)v.w;
    ((h16x4*)out)[i] = h;
  }
}

#define LOAD_CHUNK(A0, A1, Bv0, Bv1, Bv2, Bv3, KC)                              \
  do {                                                                          \
    A0  = load_A16(abase + (KC), KD_, lane);                                    \
    A1  = load_A16(abase + (size_t)16 * KD_ + (KC), KD_, lane);                 \
    Bv0 = load_B16(wbase + (KC), KD_, lane);                                    \
    Bv1 = load_B16(wbase + (size_t)16 * KD_ + (KC), KD_, lane);                 \
    Bv2 = load_B16(wbase + (size_t)32 * KD_ + (KC), KD_, lane);                 \
    Bv3 = load_B16(wbase + (size_t)48 * KD_ + (KC), KD_, lane);                 \
  } while (0)

#define DO_WMMA8(A0, A1, Bv0, Bv1, Bv2, Bv3)                                    \
  do {                                                                          \
    acc[0][0] = wmma_f16(A0, Bv0, acc[0][0]);                                   \
    acc[0][1] = wmma_f16(A0, Bv1, acc[0][1]);                                   \
    acc[0][2] = wmma_f16(A0, Bv2, acc[0][2]);                                   \
    acc[0][3] = wmma_f16(A0, Bv3, acc[0][3]);                                   \
    acc[1][0] = wmma_f16(A1, Bv0, acc[1][0]);                                   \
    acc[1][1] = wmma_f16(A1, Bv1, acc[1][1]);                                   \
    acc[1][2] = wmma_f16(A1, Bv2, acc[1][2]);                                   \
    acc[1][3] = wmma_f16(A1, Bv3, acc[1][3]);                                   \
  } while (0)

// 32(M) x 64(N) x 1024(K) wmma core, ping-pong K-chunk buffers (no rotation).
__device__ __forceinline__ void gemm_core_32x64(const _Float16* __restrict__ abase,
                                                const _Float16* __restrict__ wbase,
                                                int lane, v8f acc[2][4]) {
  v16h xa0, xa1, xb0, xb1, xb2, xb3;   // buffer X (even chunks)
  v16h ya0, ya1, yb0, yb1, yb2, yb3;   // buffer Y (odd chunks)
  LOAD_CHUNK(xa0, xa1, xb0, xb1, xb2, xb3, 0);
  for (int kc = 0; kc < KD_; kc += 64) {
    LOAD_CHUNK(ya0, ya1, yb0, yb1, yb2, yb3, kc + 32);   // always in bounds
    DO_WMMA8(xa0, xa1, xb0, xb1, xb2, xb3);
    const int k2 = (kc + 64 < KD_) ? (kc + 64) : 0;      // wrap: dummy reload
    LOAD_CHUNK(xa0, xa1, xb0, xb1, xb2, xb3, k2);
    DO_WMMA8(ya0, ya1, yb0, yb1, yb2, yb3);
  }
}

// qkv = x @ W_qkv^T + b_qkv ; scatter into Q (scaled by 1/8), K, V^T per head.
__global__ __launch_bounds__(128, 1)
void gemm_qkv_kernel(const _Float16* __restrict__ xh,
                     const _Float16* __restrict__ Wh,
                     const float* __restrict__ bias,
                     _Float16* __restrict__ Qh,
                     _Float16* __restrict__ Kh,
                     _Float16* __restrict__ Vth) {
  const int lane = threadIdx.x & 31;
  const int wave = threadIdx.x >> 5;
  const int m0 = (blockIdx.x * 4 + wave) * 32;
  const int n0 = blockIdx.y * 64;

  v8f acc[2][4] = {};
  gemm_core_32x64(xh + (size_t)m0 * KD_, Wh + (size_t)n0 * KD_, lane, acc);

  const int col = lane & 15;
  const int rh  = (lane >> 4) & 1;
#pragma unroll
  for (int j = 0; j < 4; ++j) {
    const int o    = n0 + j * 16 + col;     // output feature (row of W_qkv)
    const float bv = bias[o];
    const int head = o / 192;
    const int jj   = o % 192;               // 0..63 Q, 64..127 K, 128..191 V
#pragma unroll
    for (int mi = 0; mi < 2; ++mi) {
#pragma unroll
      for (int i = 0; i < 8; ++i) {
        const int m  = m0 + mi * 16 + i + rh * 8;   // global row = b*T + t
        const int bb = m >> 11;
        const int t  = m & (T_ - 1);
        const float v = acc[mi][j][i] + bv;
        if (jj < 64) {
          Qh[(((size_t)(bb * H_ + head)) * T_ + t) * S_ + jj] = (_Float16)(v * 0.125f);
        } else if (jj < 128) {
          Kh[(((size_t)(bb * H_ + head)) * T_ + t) * S_ + (jj - 64)] = (_Float16)v;
        } else {
          Vth[(((size_t)(bb * H_ + head)) * S_ + (jj - 128)) * (size_t)T_ + t] = (_Float16)v;
        }
      }
    }
  }
}

// Flash attention: one wave = one 16-row q tile; online softmax over k chunks of 32.
__global__ __launch_bounds__(128, 1)
void attention_kernel(const _Float16* __restrict__ Qh,
                      const _Float16* __restrict__ Kh,
                      const _Float16* __restrict__ Vth,
                      _Float16* __restrict__ resh) {
  __shared__ alignas(16) _Float16 plds[4 * 16 * 32];
  const int lane = threadIdx.x & 31;
  const int wave = threadIdx.x >> 5;
  const int bh = blockIdx.y;
  const int b  = bh >> 4;
  const int h  = bh & 15;
  const int q0 = (blockIdx.x * 4 + wave) * 16;
  _Float16* myP = plds + wave * (16 * 32);

  const _Float16* qbase = Qh + ((size_t)bh * T_ + q0) * S_;
  const v16h aq0 = load_A16(qbase, S_, lane);       // Q[:, 0:32]
  const v16h aq1 = load_A16(qbase + 32, S_, lane);  // Q[:, 32:64]

  v8f oacc[4] = {};
  float mrow[8], lrow[8];
#pragma unroll
  for (int i = 0; i < 8; ++i) { mrow[i] = -1e30f; lrow[i] = 0.0f; }

  const int col = lane & 15;
  const int rh  = (lane >> 4) & 1;
  const _Float16* kbase = Kh + (size_t)bh * T_ * S_;
  const _Float16* vbase = Vth + (size_t)bh * S_ * T_;

  // preload K tiles for chunk 0
  v16h kb0 = load_B16(kbase, S_, lane);
  v16h kb1 = load_B16(kbase + 32, S_, lane);
  v16h kb2 = load_B16(kbase + (size_t)16 * S_, S_, lane);
  v16h kb3 = load_B16(kbase + (size_t)16 * S_ + 32, S_, lane);

  for (int k0 = 0; k0 < T_; k0 += 32) {
    // scores: two 16x16 tiles (columns k0..k0+15 and k0+16..k0+31)
    v8f s0 = {}, s1 = {};
    s0 = wmma_f16(aq0, kb0, s0);
    s0 = wmma_f16(aq1, kb1, s0);
    s1 = wmma_f16(aq0, kb2, s1);
    s1 = wmma_f16(aq1, kb3, s1);

    // issue V loads (current chunk) and K loads (next chunk) so they overlap
    // the softmax VALU work below
    v16h vb0 = load_B16(vbase + k0, T_, lane);
    v16h vb1 = load_B16(vbase + (size_t)16 * T_ + k0, T_, lane);
    v16h vb2 = load_B16(vbase + (size_t)32 * T_ + k0, T_, lane);
    v16h vb3 = load_B16(vbase + (size_t)48 * T_ + k0, T_, lane);
    const int kn = (k0 + 32 < T_) ? (k0 + 32) : 0;   // wrap: harmless reload
    kb0 = load_B16(kbase + (size_t)kn * S_, S_, lane);
    kb1 = load_B16(kbase + (size_t)kn * S_ + 32, S_, lane);
    kb2 = load_B16(kbase + (size_t)(kn + 16) * S_, S_, lane);
    kb3 = load_B16(kbase + (size_t)(kn + 16) * S_ + 32, S_, lane);

#pragma unroll
    for (int i = 0; i < 8; ++i) {
      const float a = s0[i], c = s1[i];
      float mx = fmaxf(a, c);
#pragma unroll
      for (int off = 1; off < 16; off <<= 1)
        mx = fmaxf(mx, __shfl_xor(mx, off, 32));      // row max (16-lane group)
      const float mnew = fmaxf(mrow[i], mx);
      const float corr = __expf(mrow[i] - mnew);
      const float p0 = __expf(a - mnew);
      const float p1 = __expf(c - mnew);
      float rs = p0 + p1;
#pragma unroll
      for (int off = 1; off < 16; off <<= 1)
        rs += __shfl_xor(rs, off, 32);                // row sum
      lrow[i] = lrow[i] * corr + rs;
      mrow[i] = mnew;
#pragma unroll
      for (int j = 0; j < 4; ++j) oacc[j][i] *= corr;
      const int r = i + rh * 8;
      myP[r * 32 + col]      = (_Float16)p0;
      myP[r * 32 + 16 + col] = (_Float16)p1;
    }
    // P (16x32) from LDS as A-matrix; O += P @ V
    const v16h ap = load_A16(myP, 32, lane);
    oacc[0] = wmma_f16(ap, vb0, oacc[0]);
    oacc[1] = wmma_f16(ap, vb1, oacc[1]);
    oacc[2] = wmma_f16(ap, vb2, oacc[2]);
    oacc[3] = wmma_f16(ap, vb3, oacc[3]);
  }

#pragma unroll
  for (int i = 0; i < 8; ++i) {
    const float inv = 1.0f / lrow[i];
    const int r = i + rh * 8;
    const size_t row = (size_t)b * T_ + q0 + r;
#pragma unroll
    for (int j = 0; j < 4; ++j)
      resh[row * KD_ + h * S_ + j * 16 + col] = (_Float16)(oacc[j][i] * inv);
  }
}

// out = resh @ W_out^T + b_out (f32 output)
__global__ __launch_bounds__(128, 1)
void gemm_out_kernel(const _Float16* __restrict__ resh,
                     const _Float16* __restrict__ Wh,
                     const float* __restrict__ bias,
                     float* __restrict__ out) {
  const int lane = threadIdx.x & 31;
  const int wave = threadIdx.x >> 5;
  const int m0 = (blockIdx.x * 4 + wave) * 32;
  const int n0 = blockIdx.y * 64;

  v8f acc[2][4] = {};
  gemm_core_32x64(resh + (size_t)m0 * KD_, Wh + (size_t)n0 * KD_, lane, acc);

  const int col = lane & 15;
  const int rh  = (lane >> 4) & 1;
#pragma unroll
  for (int j = 0; j < 4; ++j) {
    const int n = n0 + j * 16 + col;
    const float bv = bias[n];
#pragma unroll
    for (int mi = 0; mi < 2; ++mi) {
#pragma unroll
      for (int i = 0; i < 8; ++i) {
        const int m = m0 + mi * 16 + i + rh * 8;
        out[(size_t)m * KD_ + n] = acc[mi][j][i] + bv;
      }
    }
  }
}

extern "C" void kernel_launch(void* const* d_in, const int* in_sizes, int n_in,
                              void* d_out, int out_size, void* d_ws, size_t ws_size,
                              hipStream_t stream) {
  const float* x     = (const float*)d_in[0];
  const float* W_qkv = (const float*)d_in[1];
  const float* b_qkv = (const float*)d_in[2];
  const float* W_out = (const float*)d_in[3];
  const float* b_out = (const float*)d_in[4];
  float* out = (float*)d_out;

  char* ws = (char*)d_ws;
  size_t off = 0;
  _Float16* xh    = (_Float16*)(ws + off); off += (size_t)B_ * T_ * KD_ * 2;     // 16 MB
  _Float16* Wqkvh = (_Float16*)(ws + off); off += (size_t)QKV_ * KD_ * 2;        //  6 MB
  _Float16* Wouth = (_Float16*)(ws + off); off += (size_t)KD_ * KD_ * 2;         //  2 MB
  _Float16* Qh    = (_Float16*)(ws + off); off += (size_t)B_ * H_ * T_ * S_ * 2; // 16 MB
  _Float16* Kh    = (_Float16*)(ws + off); off += (size_t)B_ * H_ * T_ * S_ * 2; // 16 MB
  _Float16* Vth   = (_Float16*)(ws + off); off += (size_t)B_ * H_ * S_ * T_ * 2; // 16 MB
  _Float16* resh  = (_Float16*)(ws + off); off += (size_t)B_ * T_ * KD_ * 2;     // 16 MB

  const int nx4  = (B_ * T_ * KD_) / 4;
  const int nw14 = (QKV_ * KD_) / 4;
  const int nw24 = (KD_ * KD_) / 4;
  f32_to_f16_kernel<<<(nx4  + 255) / 256, 256, 0, stream>>>(x,     xh,    nx4);
  f32_to_f16_kernel<<<(nw14 + 255) / 256, 256, 0, stream>>>(W_qkv, Wqkvh, nw14);
  f32_to_f16_kernel<<<(nw24 + 255) / 256, 256, 0, stream>>>(W_out, Wouth, nw24);

  // M tiles = 8192/32 = 256, 4 waves/block -> 64 blocks; N tiles of 64.
  gemm_qkv_kernel<<<dim3(64, QKV_ / 64), 128, 0, stream>>>(xh, Wqkvh, b_qkv, Qh, Kh, Vth);
  // q tiles = 128 per (b,h), 4 waves/block -> 32 blocks x; 64 (b,h) pairs y.
  attention_kernel<<<dim3(32, B_ * H_), 128, 0, stream>>>(Qh, Kh, Vth, resh);
  gemm_out_kernel<<<dim3(64, KD_ / 64), 128, 0, stream>>>(resh, Wouth, b_out, out);
}
